// UltraGCN_matrix_12575664242931
// MI455X (gfx1250) — matrix operand order repaired and checked by reading, not compiled, not run on hardware
//
#include <hip/hip_runtime.h>
#include <hip/hip_bf16.h>
#include <math.h>

// ---------------- problem constants (match reference) ----------------
#define USER_NUM 100000
#define ITEM_NUM 50000
#define HIDDEN   64
#define BATCH    16384
#define KNBR     50

#define LI_WAVES_PER_BLOCK 8
#define LI_BLOCKS  (BATCH / LI_WAVES_PER_BLOCK)   // 2048 blocks, 1 batch elem / wave
#define PAIR_BLOCK 256
#define PAIR_BLOCKS (BATCH / PAIR_BLOCK)          // 64

typedef __attribute__((ext_vector_type(16))) _Float16 v16h;
typedef __attribute__((ext_vector_type(8)))  float    v8f;

// numerically-stable log(sigmoid(x)) = min(x,0) - log1p(exp(-|x|))
__device__ __forceinline__ float log_sigmoid(float x) {
    return fminf(x, 0.0f) - log1pf(__expf(-fabsf(x)));
}

// ---------------------------------------------------------------------
// Kernel 1: per-row pos/neg logits -> LO + LC contributions.
// One thread per batch element; 128-bit loads over the 256-B rows.
// Block partial -> ws[LI_BLOCKS + blockIdx.x].
// ---------------------------------------------------------------------
__global__ void pair_kernel(const int* __restrict__ user,
                            const int* __restrict__ pos,
                            const int* __restrict__ neg,
                            const float* __restrict__ pos_beta,
                            const float* __restrict__ neg_beta,
                            const float* __restrict__ embs,
                            float* __restrict__ partial) {
    __shared__ float sred[PAIR_BLOCK];
    const int i = blockIdx.x * PAIR_BLOCK + threadIdx.x;   // i < BATCH exactly

    const float* u = embs + (size_t)user[i] * HIDDEN;
    const float* p = embs + (size_t)pos[i]  * HIDDEN;
    const float* n = embs + (size_t)neg[i]  * HIDDEN;

    float pl = 0.0f, nl = 0.0f;
#pragma unroll
    for (int j = 0; j < HIDDEN; j += 4) {
        float4 uu = *(const float4*)(u + j);
        float4 pp = *(const float4*)(p + j);
        float4 nn = *(const float4*)(n + j);
        pl += uu.x * pp.x + uu.y * pp.y + uu.z * pp.z + uu.w * pp.w;
        nl += uu.x * nn.x + uu.y * nn.y + uu.z * nn.z + uu.w * nn.w;
    }
    const float lsp = log_sigmoid(pl);       // log_sigmoid(pos_logits)
    const float lsn = log_sigmoid(-nl);      // log_sigmoid(-neg_logits)
    // LO contribution + LC contribution
    float term = -(lsp + lsn) - (lsp * pos_beta[i] + lsn * neg_beta[i]);

    sred[threadIdx.x] = term;
    __syncthreads();
#pragma unroll
    for (int s = PAIR_BLOCK / 2; s > 0; s >>= 1) {
        if ((int)threadIdx.x < s) sred[threadIdx.x] += sred[threadIdx.x + s];
        __syncthreads();
    }
    if (threadIdx.x == 0) partial[LI_BLOCKS + blockIdx.x] = sred[0];
}

// ---------------------------------------------------------------------
// Kernel 2: LI term via v_wmma_f32_16x16x32_f16.
// One wave (32 lanes) per batch element. Per element:
//   A (16x32 f16) = 16 neighbor embedding rows, one 32-dim chunk
//   B (32x16 f16) = user vector chunk broadcast into all 16 columns
//   D[m][n] = neighbor_m . user  (identical for every n)
// 4 neighbor groups x 2 K-chunks = 8 WMMAs. Column-0 lanes (0 and 16)
// read D, apply -log_sigmoid * weight. Block partial -> ws[blockIdx.x].
// ---------------------------------------------------------------------
__global__ void li_wmma_kernel(const int* __restrict__ user,
                               const int* __restrict__ pos,
                               const float* __restrict__ weights,
                               const int* __restrict__ neighbor,
                               const float* __restrict__ embs,
                               float* __restrict__ partial) {
    __shared__ float s_wave[LI_WAVES_PER_BLOCK];
    const int lane = threadIdx.x & 31;       // wave32 on gfx1250
    const int wave = threadIdx.x >> 5;
    const int half = lane >> 4;              // 0: lanes 0-15, 1: lanes 16-31
    const int r    = lane & 15;

    const int i    = blockIdx.x * LI_WAVES_PER_BLOCK + wave;   // batch element
    const int item = pos[i] - USER_NUM;                        // [0, ITEM_NUM)
    const float* urow = embs + (size_t)user[i] * HIDDEN;
    const int*   nrow = neighbor + (size_t)item * KNBR;
    const float* wrow = weights  + (size_t)item * KNBR;

    // --- B matrix (user vector), matching A's k order:
    //     lane half h holds K 8h..8h+7 (j<8) and 16+8h..16+8h+7 (j>=8); N = L % 16 (all cols equal)
    v16h bmat[2];
#pragma unroll
    for (int kc = 0; kc < 2; ++kc) {
        const float* up = urow + kc * 32 + half * 8;
#pragma unroll
        for (int j = 0; j < 16; j += 4) {
            float4 f = *(const float4*)(up + j + (j >= 8 ? 8 : 0));
            bmat[kc][j + 0] = (_Float16)f.x;
            bmat[kc][j + 1] = (_Float16)f.y;
            bmat[kc][j + 2] = (_Float16)f.z;
            bmat[kc][j + 3] = (_Float16)f.w;
        }
    }

    float li = 0.0f;

#pragma unroll
    for (int g = 0; g < 4; ++g) {                 // 4 groups of 16 neighbors (50 -> 64 padded)
        const int m   = g * 16 + r;               // neighbor index this lane's A-row feeds
        const int nid = (m < KNBR) ? nrow[m] : 0; // clamp OOB rows to row 0 (masked later)
        const float* np = embs + (size_t)nid * HIDDEN;

        v8f c = {};
#pragma unroll
        for (int kc = 0; kc < 2; ++kc) {
            // ISA 16-bit A layout (16x32): lane half=0 holds K 0-7 & 16-23 of row M=r,
            // half=1 holds K 8-15 & 24-31. Two contiguous 8-float runs -> 4x float4.
            const float* p0 = np + kc * 32 + half * 8;
            const float* p1 = p0 + 16;
            float4 f0 = *(const float4*)(p0);
            float4 f1 = *(const float4*)(p0 + 4);
            float4 f2 = *(const float4*)(p1);
            float4 f3 = *(const float4*)(p1 + 4);
            v16h a;
            a[0]  = (_Float16)f0.x; a[1]  = (_Float16)f0.y;
            a[2]  = (_Float16)f0.z; a[3]  = (_Float16)f0.w;
            a[4]  = (_Float16)f1.x; a[5]  = (_Float16)f1.y;
            a[6]  = (_Float16)f1.z; a[7]  = (_Float16)f1.w;
            a[8]  = (_Float16)f2.x; a[9]  = (_Float16)f2.y;
            a[10] = (_Float16)f2.z; a[11] = (_Float16)f2.w;
            a[12] = (_Float16)f3.x; a[13] = (_Float16)f3.y;
            a[14] = (_Float16)f3.z; a[15] = (_Float16)f3.w;

            // EXEC is all-1s here (no divergent branch encloses this call)
            c = __builtin_amdgcn_wmma_f32_16x16x32_f16(
                    /*neg_a=*/false, a, /*neg_b=*/false, bmat[kc],
                    /*c_mod=*/(short)0, c, /*reuse_a=*/false, /*reuse_b=*/false);
        }

        // D layout: lane (N = lane%16), c[q] = row M = q + 8*half. Column 0 lanes extract.
        if (r == 0) {
#pragma unroll
            for (int q = 0; q < 8; ++q) {
                const int mm = g * 16 + half * 8 + q;
                if (mm < KNBR) {
                    const float x = c[q];
                    li += -log_sigmoid(x) * wrow[mm];
                }
            }
        }
    }

    // wave reduce (only lanes 0 and 16 hold nonzero li)
    li += __shfl_xor(li, 16, 32);
    if (lane == 0) s_wave[wave] = li;
    __syncthreads();
    if (threadIdx.x == 0) {
        float acc = 0.0f;
#pragma unroll
        for (int w = 0; w < LI_WAVES_PER_BLOCK; ++w) acc += s_wave[w];
        partial[blockIdx.x] = acc;
    }
}

// ---------------------------------------------------------------------
// Kernel 3: deterministic final reduction.
// ws[0 .. LI_BLOCKS)                : LI partials (scaled by 2.5 here)
// ws[LI_BLOCKS .. LI_BLOCKS+PAIR_BLOCKS): LO+LC partials
// ---------------------------------------------------------------------
__global__ void final_kernel(const float* __restrict__ ws, float* __restrict__ out) {
    __shared__ float sred[256];
    float a = 0.0f;
    for (int j = threadIdx.x; j < LI_BLOCKS; j += 256) a += ws[j];
    float b = 0.0f;
    if ((int)threadIdx.x < PAIR_BLOCKS) b = ws[LI_BLOCKS + threadIdx.x];
    sred[threadIdx.x] = 2.5f * a + b;
    __syncthreads();
#pragma unroll
    for (int s = 128; s > 0; s >>= 1) {
        if ((int)threadIdx.x < s) sred[threadIdx.x] += sred[threadIdx.x + s];
        __syncthreads();
    }
    if (threadIdx.x == 0) out[0] = sred[0];
}

// ---------------------------------------------------------------------
extern "C" void kernel_launch(void* const* d_in, const int* in_sizes, int n_in,
                              void* d_out, int out_size, void* d_ws, size_t ws_size,
                              hipStream_t stream) {
    (void)in_sizes; (void)n_in; (void)out_size; (void)ws_size;
    const int*   user     = (const int*)d_in[0];
    const int*   pos      = (const int*)d_in[1];
    const int*   neg      = (const int*)d_in[2];
    const float* pos_beta = (const float*)d_in[3];
    const float* neg_beta = (const float*)d_in[4];
    const float* weights  = (const float*)d_in[5];
    const int*   neighbor = (const int*)d_in[6];
    const float* embs     = (const float*)d_in[7];
    float* out = (float*)d_out;
    float* ws  = (float*)d_ws;   // needs (LI_BLOCKS + PAIR_BLOCKS) floats = 8448 B

    pair_kernel<<<PAIR_BLOCKS, PAIR_BLOCK, 0, stream>>>(user, pos, neg,
                                                        pos_beta, neg_beta, embs, ws);
    li_wmma_kernel<<<LI_BLOCKS, 32 * LI_WAVES_PER_BLOCK, 0, stream>>>(user, pos, weights,
                                                                      neighbor, embs, ws);
    final_kernel<<<1, 256, 0, stream>>>(ws, out);
}
